// UnetDownAttention_17291538333702
// MI455X (gfx1250) — matrix operand order repaired and checked by reading, not compiled, not run on hardware
//
#include <hip/hip_runtime.h>
#include <math.h>

// ---------------------------------------------------------------------------
// MI455X (gfx1250) UNet-down-attention block.
// bf16 WMMA (v_wmma_f32_16x16x32_bf16) for convs (implicit GEMM), qkv,
// flash-attention (QK^T + PV), and proj. fp32 accumulation everywhere.
// Attention K/V slabs staged into LDS with async global->LDS copies
// (ASYNCcnt) shared by all 4 waves of a block, double-buffered.
// ---------------------------------------------------------------------------

typedef __attribute__((ext_vector_type(16))) __bf16 v16bf;
typedef __attribute__((ext_vector_type(8)))  __bf16 bf8;
typedef __attribute__((ext_vector_type(8)))  float  v8f;
typedef int v4i __attribute__((vector_size(16)));   // pointee type for async-LDS builtin

#define WMMA(A,B,C) __builtin_amdgcn_wmma_f32_16x16x32_bf16(false,(A),false,(B),(short)0,(C),false,false)

#define BATCH 8
#define CCH   128
#define HH    64
#define WW    64
#define TT    4096          // H*W
#define SS    4224          // CIN + T
#define NTOT  32768         // B*T
#define QSCALE 0.08838834764831845f   // 1/sqrt(128) (both q,k scales folded into q)

// ---- CDNA5 async global->LDS path (guarded; falls back to sync LDS fill) ---
#if defined(__has_builtin)
#if __has_builtin(__builtin_amdgcn_global_load_async_to_lds_b128) && \
    __has_builtin(__builtin_amdgcn_s_wait_asynccnt)
#define HAVE_ASYNC_LDS 1
#endif
#endif
#ifndef HAVE_ASYNC_LDS
#define HAVE_ASYNC_LDS 0
#endif

#define AS1 __attribute__((address_space(1)))
#define AS3 __attribute__((address_space(3)))

__device__ __forceinline__ void async_cp16(__bf16* lds_dst, const __bf16* gsrc) {
#if HAVE_ASYNC_LDS
    __builtin_amdgcn_global_load_async_to_lds_b128(
        (AS1 v4i*)gsrc, (AS3 v4i*)lds_dst, 0, 0);
#else
    *(bf8*)lds_dst = *(const bf8*)gsrc;
#endif
}

#if HAVE_ASYNC_LDS
#define WAIT_ASYNC(n) __builtin_amdgcn_s_wait_asynccnt(n)
#else
#define WAIT_ASYNC(n) ((void)0)
#endif

__device__ __forceinline__ float gelu_exact(float x) {
    return 0.5f * x * (1.0f + erff(x * 0.70710678118654752f));
}

__device__ __forceinline__ v8f vzero8() {
    v8f z;
#pragma unroll
    for (int i = 0; i < 8; ++i) z[i] = 0.0f;
    return z;
}

__device__ __forceinline__ v16bf cat8(bf8 lo, bf8 hi) {
    return __builtin_shufflevector(lo, hi, 0,1,2,3,4,5,6,7,8,9,10,11,12,13,14,15);
}

// A fragment: 16x32 bf16, row-major source, leading dim lda (elements).
// lane<16: row=lane, K 0..15; lane>=16: row=lane-16, K 16..31.
__device__ __forceinline__ v16bf load_a16(const __bf16* base, int lda) {
    int lane = threadIdx.x & 31;
    int row  = lane & 15;
    int koff = (lane >> 4) << 4;          // 0 or 16
    const __bf16* p = base + row * lda + koff;
    bf8 lo = *(const bf8*)(p);
    bf8 hi = *(const bf8*)(p + 8);
    return cat8(lo, hi);
}

// B fragment: 32x16 bf16; element (k,n) stored at base[n*ldn + k].
// lane<16: col=lane, K 0..15; lane>=16: col=lane-16, K 16..31.
__device__ __forceinline__ v16bf load_b16(const __bf16* base, int ldn) {
    int lane = threadIdx.x & 31;
    int col  = lane & 15;
    int koff = (lane >> 4) << 4;          // 0 or 16
    const __bf16* p = base + col * ldn + koff;
    bf8 lo = *(const bf8*)(p);
    bf8 hi = *(const bf8*)(p + 8);
    return cat8(lo, hi);
}

// Conv B fragment: column n -> pixel (b, h+dh, w+dw), K = channels (NHWC, contig).
__device__ __forceinline__ v16bf load_b_conv(const __bf16* xin, int n0,
                                             int dh, int dw, int ci0) {
    int lane = threadIdx.x & 31;
    int col  = lane & 15;
    int koff = (lane >> 4) << 4;
    int nn = n0 + col;
    int b  = nn >> 12;
    int t  = nn & 4095;
    int hh = (t >> 6) + dh;
    int ww = (t & 63) + dw;
    v16bf r;
    if ((unsigned)hh < 64u && (unsigned)ww < 64u) {
        const __bf16* p = xin + ((size_t)(((b << 6) + hh) << 6) + ww) * CCH + ci0 + koff;
        r = cat8(*(const bf8*)p, *(const bf8*)(p + 8));
    } else {
#pragma unroll
        for (int i = 0; i < 16; ++i) r[i] = (__bf16)0.0f;
    }
    return r;
}

// ---------------------------------------------------------------------------
// Small / elementwise kernels
// ---------------------------------------------------------------------------

// EmbedFC: emb[b][o] = W2 @ gelu(W1 @ (temb+mixemb) + b1) + b2
__global__ void emb_mlp(const float* temb, const float* mixemb,
                        const float* w1, const float* b1,
                        const float* w2, const float* b2, float* embout) {
    __shared__ float e[CCH], t1[CCH];
    int b = blockIdx.x, o = threadIdx.x;
    e[o] = temb[b * CCH + o] + mixemb[b * CCH + o];
    __syncthreads();
    float s = b1[o];
    for (int c = 0; c < CCH; ++c) s += w1[o * CCH + c] * e[c];
    t1[o] = gelu_exact(s);
    __syncthreads();
    float s2 = b2[o];
    for (int c = 0; c < CCH; ++c) s2 += w2[o * CCH + c] * t1[c];
    embout[b * CCH + o] = s2;
}

// conv weight repack: [co][ci][3][3] fp32 -> [tap][co][ci] bf16
__global__ void pack_conv(const float* w, __bf16* wp) {
    int idx = blockIdx.x * blockDim.x + threadIdx.x;   // 9*128*128
    int ci = idx & 127, co = (idx >> 7) & 127, tap = idx >> 14;
    wp[idx] = (__bf16)w[(co * CCH + ci) * 9 + tap];
}

__global__ void cvt_bf16(const float* src, __bf16* dst, int n) {
    int i = blockIdx.x * blockDim.x + threadIdx.x;
    if (i < n) dst[i] = (__bf16)src[i];
}

// xb[b][t][c] = bf16(x[b][c][t] + emb[b][c])   (NHWC for implicit-GEMM conv)
__global__ void make_xb(const float* x, const float* emb, __bf16* xb) {
    int idx = blockIdx.x * blockDim.x + threadIdx.x;   // 8*4096*128
    int c = idx & 127, t = (idx >> 7) & 4095, b = idx >> 19;
    xb[idx] = (__bf16)(x[((size_t)(b * CCH + c)) * TT + t] + emb[b * CCH + c]);
}

// per-channel batch stats over y[c][0..NTOT)
__global__ void bnstats(const float* y, float* mean, float* rstd) {
    int c = blockIdx.x;
    const float* p = y + (size_t)c * NTOT;
    float s = 0.f, s2 = 0.f;
    for (int i = threadIdx.x; i < NTOT; i += 256) { float v = p[i]; s += v; s2 += v * v; }
    __shared__ float sh[256], sh2[256];
    sh[threadIdx.x] = s; sh2[threadIdx.x] = s2;
    __syncthreads();
    for (int st = 128; st > 0; st >>= 1) {
        if (threadIdx.x < st) { sh[threadIdx.x] += sh[threadIdx.x + st];
                                sh2[threadIdx.x] += sh2[threadIdx.x + st]; }
        __syncthreads();
    }
    if (threadIdx.x == 0) {
        float m = sh[0] / (float)NTOT;
        float v = sh2[0] / (float)NTOT - m * m;
        mean[c] = m; rstd[c] = rsqrtf(v + 1e-5f);
    }
}

// x1b[b][t][c] = bf16(gelu(bn(y[c][n])))   (NHWC bf16 for conv2)
__global__ void bngelu_nhwc(const float* y, const float* mean, const float* rstd,
                            const float* g, const float* bt, __bf16* out) {
    int idx = blockIdx.x * blockDim.x + threadIdx.x;
    int c = idx & 127, t = (idx >> 7) & 4095, b = idx >> 19;
    int n = b * TT + t;
    float v = (y[(size_t)c * NTOT + n] - mean[c]) * rstd[c] * g[c] + bt[c];
    out[idx] = (__bf16)gelu_exact(v);
}

// h[b][c][t] = gelu(bn(y[c][n]))  fp32 (kept for residual + groupnorm)
__global__ void bngelu_nchw(const float* y, const float* mean, const float* rstd,
                            const float* g, const float* bt, float* h) {
    int idx = blockIdx.x * blockDim.x + threadIdx.x;  // (b*128+c)*4096+t
    int t = idx & 4095, c = (idx >> 12) & 127, b = idx >> 19;
    int n = b * TT + t;
    h[idx] = gelu_exact((y[(size_t)c * NTOT + n] - mean[c]) * rstd[c] * g[c] + bt[c]);
}

// groupnorm stats per (b, group of 16 channels): contiguous 65536 floats of h
__global__ void gnstats(const float* h, float* gmean, float* grstd) {
    int bg = blockIdx.x;                       // 64
    const float* p = h + (size_t)bg * 16 * TT; // group = 16 consecutive channels
    float s = 0.f, s2 = 0.f;
    for (int i = threadIdx.x; i < 16 * TT; i += 256) { float v = p[i]; s += v; s2 += v * v; }
    __shared__ float sh[256], sh2[256];
    sh[threadIdx.x] = s; sh2[threadIdx.x] = s2;
    __syncthreads();
    for (int st = 128; st > 0; st >>= 1) {
        if (threadIdx.x < st) { sh[threadIdx.x] += sh[threadIdx.x + st];
                                sh2[threadIdx.x] += sh2[threadIdx.x + st]; }
        __syncthreads();
    }
    if (threadIdx.x == 0) {
        float m = sh[0] / 65536.f;
        float v = sh2[0] / 65536.f - m * m;
        gmean[bg] = m; grstd[bg] = rsqrtf(v + 1e-5f);
    }
}

// hn[b][t][c] = bf16(gn(h[b][c][t]))  (c-contiguous for qkv B-fragments)
__global__ void make_hn(const float* h, const float* gmean, const float* grstd,
                        const float* g, const float* bt, __bf16* hn) {
    int idx = blockIdx.x * blockDim.x + threadIdx.x;
    int c = idx & 127, t = (idx >> 7) & 4095, b = idx >> 19;
    int bg = b * 8 + (c >> 4);
    float v = (h[((size_t)(b * CCH + c)) * TT + t] - gmean[bg]) * grstd[bg] * g[c] + bt[c];
    hn[idx] = (__bf16)v;
}

// Encoder kv via rank-1 factorization: ekv[o][j] = (enc_w[o]·mix_b) * mix_b[j] + enc_b[o]
__global__ void ek_fill(const float* mixemb, const float* enc_w, const float* enc_b,
                        __bf16* kbuf, __bf16* vbuf) {
    int b = blockIdx.x, o = threadIdx.x;   // 256 threads
    const float* mx = mixemb + b * CCH;
    float s = 0.f;
    for (int c = 0; c < CCH; ++c) s += enc_w[o * CCH + c] * mx[c];
    float bias = enc_b[o];
    if (o < CCH) {                  // ek -> k[b][s=j][c=o], j in [0,128)
        __bf16* kp = kbuf + (size_t)b * SS * CCH + o;
        for (int j = 0; j < CCH; ++j) kp[(size_t)j * CCH] = (__bf16)(s * mx[j] + bias);
    } else {                        // ev -> v[b][c=o-128][s=j]
        __bf16* vp = vbuf + ((size_t)b * CCH + (o - CCH)) * SS;
        for (int j = 0; j < CCH; ++j) vp[j] = (__bf16)(s * mx[j] + bias);
    }
}

// maxpool 2x2 on hatt[b][c][t] -> out[b][c][32][32]
__global__ void maxpool2(const float* hatt, float* out) {
    int idx = blockIdx.x * blockDim.x + threadIdx.x;  // 8*128*32*32
    int j = idx & 31, i = (idx >> 5) & 31, c = (idx >> 10) & 127, b = idx >> 17;
    const float* p = hatt + ((size_t)(b * CCH + c)) * TT + (i * 2) * WW + j * 2;
    float m = fmaxf(fmaxf(p[0], p[1]), fmaxf(p[WW], p[WW + 1]));
    out[idx] = m;
}

// ---------------------------------------------------------------------------
// WMMA GEMM kernels (4 waves/block, one 16x16 output tile per wave)
// ---------------------------------------------------------------------------

// 3x3 conv as 9 shifted K=128 GEMMs. xin NHWC bf16, wp [9][co][ci] bf16,
// y [co][B*T] fp32. Conv bias omitted: cancels exactly under batch-stat BN.
__global__ void __launch_bounds__(128) conv_wmma(const __bf16* __restrict__ xin,
                                                 const __bf16* __restrict__ wp,
                                                 float* __restrict__ y) {
    int warp = threadIdx.x >> 5, lane = threadIdx.x & 31;
    int wid = blockIdx.x * 4 + warp;
    int m0 = (wid & 7) * 16;       // co tile
    int n0 = (wid >> 3) * 16;      // spatial tile
    v8f acc = vzero8();
    for (int tap = 0; tap < 9; ++tap) {
        int dh = tap / 3 - 1, dw = tap % 3 - 1;
        const __bf16* wt = wp + tap * (CCH * CCH) + m0 * CCH;
#pragma unroll
        for (int kc = 0; kc < 4; ++kc) {
            v16bf a = load_a16(wt + kc * 32, CCH);
            v16bf b = load_b_conv(xin, n0, dh, dw, kc * 32);
            acc = WMMA(a, b, acc);
        }
    }
    int col = lane & 15, rbase = (lane >> 4) << 3;
#pragma unroll
    for (int r = 0; r < 8; ++r)
        y[(size_t)(m0 + rbase + r) * NTOT + n0 + col] = acc[r];
}

// qkv: [384x128] @ hn_b[128x4096]; scatters q (scaled, [t][c]), k ([s][c]), v ([c][s])
__global__ void __launch_bounds__(128) qkv_wmma(const __bf16* __restrict__ hn,
                                                const __bf16* __restrict__ wq,
                                                const float* __restrict__ qb,
                                                __bf16* __restrict__ qT,
                                                __bf16* __restrict__ kbuf,
                                                __bf16* __restrict__ vbuf) {
    int warp = threadIdx.x >> 5, lane = threadIdx.x & 31;
    int wid = blockIdx.x * 4 + warp;
    int b = wid / 6144, r = wid % 6144;
    int m0 = (r % 24) * 16, n0 = (r / 24) * 16;
    const __bf16* hb = hn + (size_t)b * TT * CCH;
    v8f acc = vzero8();
#pragma unroll
    for (int kc = 0; kc < 4; ++kc) {
        v16bf a = load_a16(wq + m0 * CCH + kc * 32, CCH);
        v16bf bb = load_b16(hb + (size_t)n0 * CCH + kc * 32, CCH);
        acc = WMMA(a, bb, acc);
    }
    int col = lane & 15, rbase = (lane >> 4) << 3;
#pragma unroll
    for (int r8 = 0; r8 < 8; ++r8) {
        int o = m0 + rbase + r8, t = n0 + col;
        float val = acc[r8] + qb[o];
        if (o < CCH)
            qT[((size_t)(b * TT + t)) * CCH + o] = (__bf16)(val * QSCALE);
        else if (o < 2 * CCH)
            kbuf[((size_t)b * SS + CCH + t) * CCH + (o - CCH)] = (__bf16)val;
        else
            vbuf[((size_t)b * CCH + (o - 2 * CCH)) * SS + CCH + t] = (__bf16)val;
    }
}

// Stage one 32-wide K/V slab into LDS (cooperative across the 128-thread block,
// 8 async b128 instructions per wave, tracked by ASYNCcnt).
__device__ __forceinline__ void stage_kv(__bf16* kdst, __bf16* vdst,
                                         const __bf16* kb, const __bf16* vb, int s0) {
    int tid = threadIdx.x;
    // K slab: kbuf[s0..s0+32)[0..128) is one contiguous 8KB block -> flat copy
    const __bf16* gk = kb + (size_t)s0 * CCH;
#pragma unroll
    for (int j = 0; j < 4; ++j) {
        int off = (tid + j * 128) * 8;
        async_cp16(kdst + off, gk + off);
    }
    // V slab: row c = tid (stride SS in global), 32 bf16 -> 4 chunks of 16B
    const __bf16* gv = vb + (size_t)tid * SS + s0;
#pragma unroll
    for (int j = 0; j < 4; ++j)
        async_cp16(vdst + tid * 32 + j * 8, gv + j * 8);
}

// Flash attention: each wave = 16 query rows, streams S in 32-wide slabs.
// K/V slabs double-buffered in LDS, filled by async global->LDS copies and
// shared by all 4 waves (they read identical K/V, distinct query tiles).
__global__ void __launch_bounds__(128) attn_wmma(const __bf16* __restrict__ qT,
                                                 const __bf16* __restrict__ kbuf,
                                                 const __bf16* __restrict__ vbuf,
                                                 __bf16* __restrict__ abuf) {
    __shared__ __bf16 ksl[2][32 * CCH];     // [s_local][c]  8KB each
    __shared__ __bf16 vsl[2][CCH * 32];     // [c][s_local]  8KB each
    __shared__ float  sc[4][16][32];
    __shared__ __bf16 pb[4][16][32];
    __shared__ float  rowm[4][16], rows[4][16], corr[4][16];
    int warp = threadIdx.x >> 5, lane = threadIdx.x & 31;
    int wid = blockIdx.x * 4 + warp;
    int b = wid >> 8;                       // 4 waves/block never straddle b
    int t0 = (wid & 255) << 4;
    const __bf16* q  = qT + ((size_t)(b * TT + t0)) * CCH;
    const __bf16* kb = kbuf + (size_t)b * SS * CCH;
    const __bf16* vb = vbuf + (size_t)b * CCH * SS;
    if (lane < 16) { rowm[warp][lane] = -3.0e38f; rows[warp][lane] = 0.0f; }
    v8f acc[8];
#pragma unroll
    for (int ct = 0; ct < 8; ++ct) acc[ct] = vzero8();
    // q fragments are s-loop invariant: hoist
    v16bf qa[4];
#pragma unroll
    for (int kc = 0; kc < 4; ++kc) qa[kc] = load_a16(q + kc * 32, CCH);
    int col = lane & 15, rbase = (lane >> 4) << 3;

    stage_kv(ksl[0], vsl[0], kb, vb, 0);    // prefetch slab 0
    __syncthreads();

    for (int it = 0; it < SS / 32; ++it) {
        int s0 = it * 32;
        int cur = it & 1;
        if (s0 + 32 < SS) {                 // prefetch next slab into other buffer
            stage_kv(ksl[cur ^ 1], vsl[cur ^ 1], kb, vb, s0 + 32);
            WAIT_ASYNC(8);                  // drain the 8 older copies (in-order)
        } else {
            WAIT_ASYNC(0);
        }
        __syncthreads();                    // LDS slab visible block-wide

        // scores S[t0..+16][s0..+32] = q . k  (K = 128 channels, from LDS)
#pragma unroll
        for (int hf = 0; hf < 2; ++hf) {
            v8f s = vzero8();
#pragma unroll
            for (int kc = 0; kc < 4; ++kc) {
                v16bf bb = load_b16(&ksl[cur][(hf * 16) * CCH + kc * 32], CCH);
                s = WMMA(qa[kc], bb, s);
            }
#pragma unroll
            for (int r = 0; r < 8; ++r) sc[warp][rbase + r][hf * 16 + col] = s[r];
        }
        __syncthreads();
        // online softmax (lane l handles row l)
        if (lane < 16) {
            float mo = rowm[warp][lane], mx = mo;
#pragma unroll
            for (int j = 0; j < 32; ++j) mx = fmaxf(mx, sc[warp][lane][j]);
            float cf = __expf(mo - mx);
            float sum = rows[warp][lane] * cf;
#pragma unroll
            for (int j = 0; j < 32; ++j) {
                float p = __expf(sc[warp][lane][j] - mx);
                pb[warp][lane][j] = (__bf16)p;
                sum += p;
            }
            rowm[warp][lane] = mx; rows[warp][lane] = sum; corr[warp][lane] = cf;
        }
        __syncthreads();
        // rescale accumulators by per-row correction
#pragma unroll
        for (int r = 0; r < 8; ++r) {
            float cf = corr[warp][rbase + r];
#pragma unroll
            for (int ct = 0; ct < 8; ++ct) acc[ct][r] *= cf;
        }
        // PV: A = probs (16x32 from LDS), B = v slabs (LDS, [c][s_local])
        v16bf pa = load_a16(&pb[warp][0][0], 32);
#pragma unroll
        for (int ct = 0; ct < 8; ++ct) {
            v16bf bb = load_b16(&vsl[cur][(ct * 16) * 32], 32);
            acc[ct] = WMMA(pa, bb, acc[ct]);
        }
        __syncthreads();                    // done with slab `cur` before reuse
    }
    // normalize and emit a[t][c]
#pragma unroll
    for (int r = 0; r < 8; ++r) {
        float inv = 1.0f / rows[warp][rbase + r];
        int t = t0 + rbase + r;
#pragma unroll
        for (int ct = 0; ct < 8; ++ct)
            abuf[((size_t)(b * TT + t)) * CCH + ct * 16 + col] = (__bf16)(acc[ct][r] * inv);
    }
}

// proj + residual: hatt[b][o][t] = h[b][o][t] + proj_w@a + proj_b
__global__ void __launch_bounds__(128) proj_wmma(const __bf16* __restrict__ abuf,
                                                 const __bf16* __restrict__ wp,
                                                 const float* __restrict__ pbias,
                                                 const float* __restrict__ h,
                                                 float* __restrict__ hatt) {
    int warp = threadIdx.x >> 5, lane = threadIdx.x & 31;
    int wid = blockIdx.x * 4 + warp;
    int b = wid >> 11, r = wid & 2047;
    int m0 = (r & 7) * 16, n0 = (r >> 3) * 16;
    v8f acc = vzero8();
#pragma unroll
    for (int kc = 0; kc < 4; ++kc) {
        v16bf a  = load_a16(wp + m0 * CCH + kc * 32, CCH);
        v16bf bb = load_b16(abuf + ((size_t)(b * TT + n0)) * CCH + kc * 32, CCH);
        acc = WMMA(a, bb, acc);
    }
    int col = lane & 15, rbase = (lane >> 4) << 3;
#pragma unroll
    for (int r8 = 0; r8 < 8; ++r8) {
        int o = m0 + rbase + r8, t = n0 + col;
        size_t idx = ((size_t)(b * CCH + o)) * TT + t;
        hatt[idx] = h[idx] + acc[r8] + pbias[o];
    }
}

// ---------------------------------------------------------------------------

extern "C" void kernel_launch(void* const* d_in, const int* in_sizes, int n_in,
                              void* d_out, int out_size, void* d_ws, size_t ws_size,
                              hipStream_t stream) {
    (void)in_sizes; (void)n_in; (void)out_size; (void)ws_size;
    const float* x       = (const float*)d_in[0];
    const float* temb    = (const float*)d_in[1];
    const float* mixemb  = (const float*)d_in[2];
    const float* emb_w1  = (const float*)d_in[3];
    const float* emb_b1  = (const float*)d_in[4];
    const float* emb_w2  = (const float*)d_in[5];
    const float* emb_b2  = (const float*)d_in[6];
    const float* conv1_w = (const float*)d_in[7];
    const float* bn1_g   = (const float*)d_in[9];
    const float* bn1_b   = (const float*)d_in[10];
    const float* conv2_w = (const float*)d_in[11];
    const float* bn2_g   = (const float*)d_in[13];
    const float* bn2_b   = (const float*)d_in[14];
    const float* gn_g    = (const float*)d_in[15];
    const float* gn_b    = (const float*)d_in[16];
    const float* qkv_w   = (const float*)d_in[17];
    const float* qkv_b   = (const float*)d_in[18];
    const float* enc_w   = (const float*)d_in[19];
    const float* enc_b   = (const float*)d_in[20];
    const float* proj_w  = (const float*)d_in[21];
    const float* proj_b  = (const float*)d_in[22];
    float* out = (float*)d_out;

    // workspace carve-up (256B aligned); ~77 MB total with reuse
    char* base = (char*)d_ws;
    size_t off = 0;
    auto carve = [&](size_t bytes) {
        void* p = base + off;
        off = (off + bytes + 255) & ~(size_t)255;
        return p;
    };
    float*  emb   = (float*) carve(BATCH * CCH * 4);
    float*  mean1 = (float*) carve(CCH * 4);
    float*  rstd1 = (float*) carve(CCH * 4);
    float*  mean2 = (float*) carve(CCH * 4);
    float*  rstd2 = (float*) carve(CCH * 4);
    float*  gmean = (float*) carve(64 * 4);
    float*  grstd = (float*) carve(64 * 4);
    __bf16* wp1   = (__bf16*)carve(9 * CCH * CCH * 2);
    __bf16* wp2   = (__bf16*)carve(9 * CCH * CCH * 2);
    __bf16* wq    = (__bf16*)carve(3 * CCH * CCH * 2);
    __bf16* wproj = (__bf16*)carve(CCH * CCH * 2);
    __bf16* xb    = (__bf16*)carve((size_t)BATCH * TT * CCH * 2);  // later reused as hn
    float*  ybuf  = (float*) carve((size_t)CCH * NTOT * 4);        // later reused as hatt
    __bf16* x1b   = (__bf16*)carve((size_t)BATCH * TT * CCH * 2);  // later reused as abuf
    float*  hbuf  = (float*) carve((size_t)BATCH * CCH * TT * 4);
    __bf16* qT    = (__bf16*)carve((size_t)BATCH * TT * CCH * 2);
    __bf16* kbuf  = (__bf16*)carve((size_t)BATCH * SS * CCH * 2);
    __bf16* vbuf  = (__bf16*)carve((size_t)BATCH * CCH * SS * 2);
    __bf16* hn    = xb;
    __bf16* abuf  = x1b;
    float*  hatt  = ybuf;

    const int EW_GRID = (BATCH * TT * CCH) / 256;    // 16384

    // 1) conditioning MLP
    emb_mlp<<<BATCH, CCH, 0, stream>>>(temb, mixemb, emb_w1, emb_b1, emb_w2, emb_b2, emb);
    // 2) weight packs
    pack_conv<<<(9 * CCH * CCH) / 256, 256, 0, stream>>>(conv1_w, wp1);
    pack_conv<<<(9 * CCH * CCH) / 256, 256, 0, stream>>>(conv2_w, wp2);
    cvt_bf16<<<(3 * CCH * CCH) / 256, 256, 0, stream>>>(qkv_w, wq, 3 * CCH * CCH);
    cvt_bf16<<<(CCH * CCH) / 256, 256, 0, stream>>>(proj_w, wproj, CCH * CCH);
    // 3) x + emb -> NHWC bf16
    make_xb<<<EW_GRID, 256, 0, stream>>>(x, emb, xb);
    // 4-6) conv1 -> BN stats -> BN+GELU (bias skipped: cancels under BN)
    conv_wmma<<<4096, 128, 0, stream>>>(xb, wp1, ybuf);
    bnstats<<<CCH, 256, 0, stream>>>(ybuf, mean1, rstd1);
    bngelu_nhwc<<<EW_GRID, 256, 0, stream>>>(ybuf, mean1, rstd1, bn1_g, bn1_b, x1b);
    // 7-9) conv2 -> BN stats -> BN+GELU (h kept fp32 NCHW)
    conv_wmma<<<4096, 128, 0, stream>>>(x1b, wp2, ybuf);
    bnstats<<<CCH, 256, 0, stream>>>(ybuf, mean2, rstd2);
    bngelu_nchw<<<EW_GRID, 256, 0, stream>>>(ybuf, mean2, rstd2, bn2_g, bn2_b, hbuf);
    // 10-11) groupnorm -> hn[t][c] bf16
    gnstats<<<64, 256, 0, stream>>>(hbuf, gmean, grstd);
    make_hn<<<EW_GRID, 256, 0, stream>>>(hbuf, gmean, grstd, gn_g, gn_b, hn);
    // 12) encoder k/v via rank-1 factorization of enc_w @ (mix ⊗ mix)
    ek_fill<<<BATCH, 256, 0, stream>>>(mixemb, enc_w, enc_b, kbuf, vbuf);
    // 13) qkv projection (q pre-scaled by 1/sqrt(C))
    qkv_wmma<<<12288, 128, 0, stream>>>(hn, wq, qkv_b, qT, kbuf, vbuf);
    // 14) flash attention over S=4224 (async LDS-staged K/V)
    attn_wmma<<<512, 128, 0, stream>>>(qT, kbuf, vbuf, abuf);
    // 15) proj + residual
    proj_wmma<<<4096, 128, 0, stream>>>(abuf, wproj, proj_b, hbuf, hatt);
    // 16) maxpool 2x2 -> out [8,128,32,32]
    maxpool2<<<(BATCH * CCH * 32 * 32) / 256, 256, 0, stream>>>(hatt, out);
}
